// AttnDecoderGRU_40209483825413
// MI455X (gfx1250) — compile-verified, matrix operand order checked by reading, not run on hardware
//
#include <hip/hip_runtime.h>
#include <hip/hip_bf16.h>
#include <math.h>

// ---------------------------------------------------------------------------
// Types for CDNA5 WMMA (wave32): D(16x16 f32) = A(16x32 bf16) x B(32x16 bf16) + C
// ---------------------------------------------------------------------------
typedef __attribute__((ext_vector_type(16))) __bf16    v16bf;
typedef __attribute__((ext_vector_type(8)))  float     v8f;
typedef __attribute__((ext_vector_type(8)))  unsigned  v8u;

// Pack two fp32 -> bf16x2 dword with a single v_perm_b32 (truncation).
__device__ __forceinline__ unsigned pack2(float lo, float hi) {
    return __builtin_amdgcn_perm(__builtin_bit_cast(unsigned, hi),
                                 __builtin_bit_cast(unsigned, lo),
                                 0x07060302u);
}

// Raw fp32 fragment: 8 contiguous floats at +c0 and 8 at +c1 (four b128 loads).
struct Raw { float4 q0, q1, q2, q3; };

__device__ __forceinline__ Raw load_raw(const float* __restrict__ p, int c0, int c1) {
    Raw r;
    r.q0 = *(const float4*)(p + c0);
    r.q1 = *(const float4*)(p + c0 + 4);
    r.q2 = *(const float4*)(p + c1);
    r.q3 = *(const float4*)(p + c1 + 4);
    return r;
}

// Convert raw fp32 fragment -> bf16 fragment (8 v_perm_b32).
// Matches CDNA5 16-bit A(16x32)/B(32x16) per-lane layout: VGPR v holds
// K = {c0+2v, c0+2v+1} for v<4 and {c1+2(v-4), ...} for v>=4.
__device__ __forceinline__ v16bf cvt_frag(const Raw& r) {
    v8u u;
    u[0] = pack2(r.q0.x, r.q0.y); u[1] = pack2(r.q0.z, r.q0.w);
    u[2] = pack2(r.q1.x, r.q1.y); u[3] = pack2(r.q1.z, r.q1.w);
    u[4] = pack2(r.q2.x, r.q2.y); u[5] = pack2(r.q2.z, r.q2.w);
    u[6] = pack2(r.q3.x, r.q3.y); u[7] = pack2(r.q3.z, r.q3.w);
    return __builtin_bit_cast(v16bf, u);
}

#define WMMA_BF16(Af, Bf, Cf) \
    __builtin_amdgcn_wmma_f32_16x16x32_bf16(false, (Af), false, (Bf), (short)0, (Cf), false, false)

// ---------------------------------------------------------------------------
// Generic NT GEMM: C[M,N] = A[M,K] * B[N,K]^T (+bias[N]) (optional tanh)
// A, B fp32 row-major in HBM, converted to bf16 in registers.
// One wave -> 16x32 output strip (2 accumulators; 1 A + 2 B fragments/step).
// Tile index: tm varies FASTEST so concurrent waves covering all of M share
// the same B rows -> the big weight matrix streams from HBM exactly once.
// K-loop software-pipelined with NAMED ping-pong buffers (manual x2 unroll).
// ---------------------------------------------------------------------------
__global__ void gemm_nt_bf16wmma(const float* __restrict__ A, int lda,
                                 const float* __restrict__ Bm, int ldb,
                                 const float* __restrict__ bias,
                                 float* __restrict__ C, int ldc,
                                 int M, int N, int K, int act) {
    const int ntm   = M >> 4;
    const int tiles = ntm * (N >> 5);
    const int wave  = blockIdx.x * (blockDim.x >> 5) + (threadIdx.x >> 5);
    if (wave >= tiles) return;
    const int tm = wave % ntm;          // fast index: share B across M-tiles
    const int tn = wave / ntm;
    const int m0 = tm << 4;
    const int n0 = tn << 5;

    const int lane   = threadIdx.x & 31;
    const int laneLo = lane & 15;
    const int hi     = lane >> 4;       // lane half selects K sub-chunk
    const int h8     = hi * 8;

    const float* __restrict__ arow = A  + (size_t)(m0 + laneLo) * lda;
    const float* __restrict__ br0  = Bm + (size_t)(n0 + laneLo) * ldb;
    const float* __restrict__ br1  = br0 + (size_t)16 * ldb;

    v8f acc0 = {}, acc1 = {};

    // ping-pong raw buffers (named scalars -> guaranteed VGPRs)
    Raw a0, b00, b10, a1, b01, b11;
    a0  = load_raw(arow, h8, 16 + h8);
    b00 = load_raw(br0,  h8, 16 + h8);
    b10 = load_raw(br1,  h8, 16 + h8);

    // K is a multiple of 64 for all our shapes -> process two 32-steps/iter
    for (int k0 = 0; k0 < K; k0 += 64) {
        {   // prefetch k0+32 into buffer set 1
            const int c0 = k0 + 32 + h8, c1 = k0 + 48 + h8;
            a1  = load_raw(arow, c0, c1);
            b01 = load_raw(br0,  c0, c1);
            b11 = load_raw(br1,  c0, c1);
        }
        {   // compute on buffer set 0 (k0)
            v16bf af = cvt_frag(a0);
            v16bf f0 = cvt_frag(b00);
            v16bf f1 = cvt_frag(b10);
            acc0 = WMMA_BF16(af, f0, acc0);
            acc1 = WMMA_BF16(af, f1, acc1);
        }
        if (k0 + 64 < K) {  // prefetch k0+64 into buffer set 0
            const int c0 = k0 + 64 + h8, c1 = k0 + 80 + h8;
            a0  = load_raw(arow, c0, c1);
            b00 = load_raw(br0,  c0, c1);
            b10 = load_raw(br1,  c0, c1);
        }
        {   // compute on buffer set 1 (k0+32)
            v16bf af = cvt_frag(a1);
            v16bf f0 = cvt_frag(b01);
            v16bf f1 = cvt_frag(b11);
            acc0 = WMMA_BF16(af, f0, acc0);
            acc1 = WMMA_BF16(af, f1, acc1);
        }
    }

    // C/D layout: element r, lane L -> row m0 + r + 8*(L>=16), col n0 + (L&15)
    v8f accs[2] = {acc0, acc1};
    #pragma unroll
    for (int j = 0; j < 2; ++j) {
        const int col = n0 + 16 * j + laneLo;
        const float bv = bias ? bias[col] : 0.f;
        #pragma unroll
        for (int r = 0; r < 8; ++r) {
            float v = accs[j][r] + bv;
            if (act == 1) v = tanhf(v);
            C[(size_t)(m0 + r + 8 * hi) * ldc + col] = v;
        }
    }
}

// ---------------------------------------------------------------------------
// Elementwise kernels
// ---------------------------------------------------------------------------
__global__ void embed_gather(const int* __restrict__ idx, const float* __restrict__ emb,
                             float* __restrict__ x, int Bn, int Hn) {
    int i = blockIdx.x * blockDim.x + threadIdx.x;
    if (i >= Bn * Hn) return;
    int b = i / Hn, h = i - b * Hn;
    x[i] = emb[(size_t)idx[b] * Hn + h];
}

__device__ __forceinline__ float sigmf(float x) { return 1.f / (1.f + expf(-x)); }

__global__ void gru_gates(const float* __restrict__ gi, const float* __restrict__ gh,
                          const float* __restrict__ hprev,
                          float* __restrict__ hout, float* __restrict__ cat,
                          int Bn, int Hn) {
    int i = blockIdx.x * blockDim.x + threadIdx.x;
    if (i >= Bn * Hn) return;
    int b = i / Hn, j = i - b * Hn;
    const float* gib = gi + (size_t)b * 3 * Hn;
    const float* ghb = gh + (size_t)b * 3 * Hn;
    float r = sigmf(gib[j] + ghb[j]);
    float z = sigmf(gib[Hn + j] + ghb[Hn + j]);
    float n = tanhf(gib[2 * Hn + j] + r * ghb[2 * Hn + j]);
    float hn = (1.f - z) * n + z * hprev[i];
    hout[i] = hn;                               // new hidden (output #2)
    cat[(size_t)b * 2 * Hn + j] = hn;           // [h_new | context] staging
}

// ---------------------------------------------------------------------------
// Fused Luong attention: one block per batch row b.
//   pass 1: scores[s] = dot(h_new[b,:], enc[s,b,:])   (streams 768KB from HBM)
//   softmax over S in LDS
//   pass 2: context[h] = sum_s attn[s]*enc[s,b,h]     (same 768KB -> L2 hit)
// Halves encoder_outputs HBM traffic vs separate kernels (~200MB saved).
// ---------------------------------------------------------------------------
__global__ void attn_fused(const float* __restrict__ hnew, const float* __restrict__ enc,
                           float* __restrict__ cat, int Bn, int Sn, int Hn) {
    __shared__ float sc[128];
    __shared__ float red[128];
    const int b = blockIdx.x;
    const int tid = threadIdx.x;
    const int waveId = tid >> 5, lane = tid & 31;
    const float* hp = hnew + (size_t)b * Hn;

    // pass 1: one wave per score row, lane-strided over H
    for (int s = waveId; s < Sn; s += 8) {
        const float* ep = enc + ((size_t)s * Bn + b) * Hn;
        float acc = 0.f;
        for (int h = lane; h < Hn; h += 32) acc += hp[h] * ep[h];
        #pragma unroll
        for (int m = 16; m; m >>= 1) acc += __shfl_xor(acc, m, 32);
        if (lane == 0) sc[s] = acc;
    }
    __syncthreads();

    // softmax over S=128 (threads 0..127 participate in reductions)
    if (tid < Sn) red[tid] = sc[tid];
    __syncthreads();
    for (int off = 64; off; off >>= 1) { if (tid < off) red[tid] = fmaxf(red[tid], red[tid + off]); __syncthreads(); }
    const float mx = red[0];
    __syncthreads();
    if (tid < Sn) { float e = expf(sc[tid] - mx); sc[tid] = e; red[tid] = e; }
    __syncthreads();
    for (int off = 64; off; off >>= 1) { if (tid < off) red[tid] += red[tid + off]; __syncthreads(); }
    const float inv = 1.f / red[0];
    __syncthreads();
    if (tid < Sn) sc[tid] *= inv;
    __syncthreads();

    // pass 2: context, coalesced over h; enc slice re-read hits L2
    for (int h = tid; h < Hn; h += 256) {
        const float* eb = enc + (size_t)b * Hn + h;
        float acc = 0.f;
        for (int s = 0; s < Sn; ++s) acc += sc[s] * eb[(size_t)s * Bn * Hn];
        cat[((size_t)b * 2 + 1) * Hn + h] = acc;
    }
}

__global__ void softmax_v(float* __restrict__ logits, int Vn) {
    __shared__ float red[256];
    int b = blockIdx.x, t = threadIdx.x;
    float* row = logits + (size_t)b * Vn;
    float mx = -INFINITY;
    for (int v = t; v < Vn; v += 256) mx = fmaxf(mx, row[v]);
    red[t] = mx; __syncthreads();
    for (int off = 128; off; off >>= 1) { if (t < off) red[t] = fmaxf(red[t], red[t + off]); __syncthreads(); }
    mx = red[0]; __syncthreads();
    float sum = 0.f;
    for (int v = t; v < Vn; v += 256) sum += expf(row[v] - mx);
    red[t] = sum; __syncthreads();
    for (int off = 128; off; off >>= 1) { if (t < off) red[t] += red[t + off]; __syncthreads(); }
    float inv = 1.f / red[0];
    for (int v = t; v < Vn; v += 256) row[v] = expf(row[v] - mx) * inv;
}

// ---------------------------------------------------------------------------
// Host side
// ---------------------------------------------------------------------------
extern "C" void kernel_launch(void* const* d_in, const int* in_sizes, int n_in,
                              void* d_out, int out_size, void* d_ws, size_t ws_size,
                              hipStream_t stream) {
    constexpr int Bsz = 256, Hsz = 1536, Vsz = 32000, Ssz = 128;

    const int*   input_step  = (const int*)  d_in[0];
    const float* last_hidden = (const float*)d_in[1];   // [1,B,H] == [B,H]
    const float* enc         = (const float*)d_in[2];   // [S,B,H]
    const float* emb         = (const float*)d_in[3];   // [V,H]
    const float* W_ih        = (const float*)d_in[4];   // [3H,H]
    const float* W_hh        = (const float*)d_in[5];   // [3H,H]
    const float* b_ih        = (const float*)d_in[6];
    const float* b_hh        = (const float*)d_in[7];
    const float* W_concat    = (const float*)d_in[8];   // [H,2H]
    const float* b_concat    = (const float*)d_in[9];
    const float* W_out       = (const float*)d_in[10];  // [V,H]
    const float* b_out       = (const float*)d_in[11];

    // Workspace layout (floats), ~15.8 MB total
    float* ws    = (float*)d_ws;
    float* x     = ws;                           // [B,H]   embedded input
    float* gi    = x    + (size_t)Bsz * Hsz;     // [B,3H]
    float* gh    = gi   + (size_t)Bsz * 3 * Hsz; // [B,3H]
    float* cat   = gh   + (size_t)Bsz * 3 * Hsz; // [B,2H] = [h_new | context]
    float* cout  = cat  + (size_t)Bsz * 2 * Hsz; // [B,H]  tanh(concat proj)

    float* probs = (float*)d_out;                // [B,V]
    float* hout  = probs + (size_t)Bsz * Vsz;    // [1,B,H]

    auto gemm = [&](const float* A, int lda, const float* Bm, int ldb,
                    const float* bias, float* C, int ldc,
                    int M, int N, int K, int act) {
        int tiles  = (M / 16) * (N / 32);
        int blocks = (tiles + 7) / 8;            // 8 waves / block (wave32)
        gemm_nt_bf16wmma<<<blocks, 256, 0, stream>>>(A, lda, Bm, ldb, bias, C, ldc, M, N, K, act);
    };

    // 1) embedding gather
    embed_gather<<<(Bsz * Hsz + 255) / 256, 256, 0, stream>>>(input_step, emb, x, Bsz, Hsz);
    // 2) GRU input/hidden gate pre-activations (bf16 WMMA, fp32 accumulate)
    gemm(x,           Hsz, W_ih, Hsz, b_ih, gi, 3 * Hsz, Bsz, 3 * Hsz, Hsz, 0);
    gemm(last_hidden, Hsz, W_hh, Hsz, b_hh, gh, 3 * Hsz, Bsz, 3 * Hsz, Hsz, 0);
    // 3) gates -> h_new (output #2) and cat[:, :H]
    gru_gates<<<(Bsz * Hsz + 255) / 256, 256, 0, stream>>>(gi, gh, last_hidden, hout, cat, Bsz, Hsz);
    // 4) fused attention: scores + softmax + context, enc read ~once from HBM
    attn_fused<<<Bsz, 256, 0, stream>>>(hout, enc, cat, Bsz, Ssz, Hsz);
    // 5) concat projection with fused tanh (bf16 WMMA)
    gemm(cat,  2 * Hsz, W_concat, 2 * Hsz, b_concat, cout, Hsz, Bsz, Hsz, 2 * Hsz, 1);
    // 6) output projection -> logits in d_out (bf16 WMMA; 197 MB weight stream)
    gemm(cout, Hsz, W_out, Hsz, b_out, probs, Vsz, Bsz, Vsz, Hsz, 0);
    // 7) softmax over V in place
    softmax_v<<<Bsz, 256, 0, stream>>>(probs, Vsz);
}
